// GCNDecoder_28432683500206
// MI455X (gfx1250) — matrix operand order repaired and checked by reading, not compiled, not run on hardware
//
#include <hip/hip_runtime.h>
#include <hip/hip_bf16.h>
#include <stdint.h>
#include <stddef.h>

// Problem constants (from the reference)
#define N_NODES 50000
#define N_EDGES 800000
#define DIM_IN  256
#define DIM_H1  512
#define DIM_H2  1024
#define DIM_OUT 3000
#define EPSV    1e-5f

typedef __attribute__((ext_vector_type(16))) __bf16       v16bf;
typedef __attribute__((ext_vector_type(8)))  float        v8f;
typedef __attribute__((ext_vector_type(4)))  unsigned int u32x4;

// ---------------------------------------------------------------------------
// Small elementwise / graph-prep kernels
// ---------------------------------------------------------------------------

// W (K x Nt, row-major f32)  ->  Wt (Nt x K, row-major bf16)  [transpose+convert]
__global__ void k_cvt_wt(const float* __restrict__ W, __bf16* __restrict__ Wt,
                         int K, int Nt) {
  int idx = blockIdx.x * blockDim.x + threadIdx.x;
  if (idx >= K * Nt) return;
  int k = idx / Nt;
  int n = idx - k * Nt;
  Wt[(size_t)n * K + k] = (__bf16)W[idx];
}

__global__ void k_cvt_act(const float* __restrict__ X, __bf16* __restrict__ Xb,
                          size_t total) {
  size_t i = (size_t)blockIdx.x * blockDim.x + threadIdx.x;
  if (i < total) Xb[i] = (__bf16)X[i];
}

__global__ void k_deg(const int* __restrict__ ei, const float* __restrict__ ea,
                      float* __restrict__ deg) {
  int e = blockIdx.x * blockDim.x + threadIdx.x;
  if (e >= N_EDGES) return;
  int dst = ei[N_EDGES + e];
  atomicAdd(&deg[dst], ea[3 * e + 2]);
}

__global__ void k_dinv(const float* __restrict__ deg, float* __restrict__ dinv,
                       float* __restrict__ dinv2) {
  int i = blockIdx.x * blockDim.x + threadIdx.x;
  if (i >= N_NODES) return;
  float d = rsqrtf(deg[i] + 1.0f);
  dinv[i]  = d;
  dinv2[i] = d * d;
}

__global__ void k_coef(const int* __restrict__ ei, const float* __restrict__ ea,
                       const float* __restrict__ dinv, float* __restrict__ coef) {
  int e = blockIdx.x * blockDim.x + threadIdx.x;
  if (e >= N_EDGES) return;
  coef[e] = dinv[ei[e]] * ea[3 * e + 2] * dinv[ei[N_EDGES + e]];
}

// ---------------------------------------------------------------------------
// bf16 WMMA GEMM:  C[M x Nt] (f32) = A[M x K] (bf16, row-major)
//                                  @ Bt[Nt x K] (bf16, N-major / pre-transposed)
//
// Block: 256 threads (8 wave32), tile 128(M) x 128(N) x 32(K).
// Waves arranged 4(M) x 2(N); each wave owns a 32x64 patch = 2x4
// v_wmma_f32_16x16x32_bf16 accumulators.
//
// Data movement uses the CDNA5 async path: GLOBAL_LOAD_ASYNC_TO_LDS_B128
// (ASYNCcnt) with double-buffered LDS tiles. OOB rows/cols are *clamped*
// (not guarded) so every wave issues exactly 4 async ops per tile and the
// s_wait_asynccnt thresholds stay exact; clamped data only ever feeds
// output elements that the store-guard drops.
// ---------------------------------------------------------------------------
#define BM 128
#define BN 128
#define BK 32
#define LDP 40   // padded LDS row (bf16 elements); 80B rows keep 16B alignment

__global__ __launch_bounds__(256)
void k_gemm_bf16(const __bf16* __restrict__ A, const __bf16* __restrict__ Bt,
                 float* __restrict__ C, int M, int Nt, int K) {
  __shared__ __align__(16) __bf16 As[2][BM][LDP];
  __shared__ __align__(16) __bf16 Bs[2][BN][LDP];

  const int tid  = threadIdx.x;
  const int lane = tid & 31;
  const int wave = tid >> 5;
  const int wm   = (wave & 3) * 32;   // wave M offset (4 waves along M)
  const int wn   = (wave >> 2) * 64;  // wave N offset (2 waves along N)
  const int lr   = lane & 15;
  const int half = lane >> 4;

  const int m0 = blockIdx.x * BM;
  const int n0 = blockIdx.y * BN;

  // global->LDS staging: each thread owns one A row-chunk and one B row-chunk
  const int arow = tid >> 1;            // 0..127
  const int ach  = (tid & 1) * 16;      // col chunk base (16 bf16 = 32B, 2 ops)
  const int ga   = m0 + arow;
  const int gb   = n0 + arow;
  const int gac  = ga < M  ? ga : (M - 1);   // clamp: keeps EXEC full + counters uniform
  const int gbc  = gb < Nt ? gb : (Nt - 1);

  const __bf16* gA = A  + (size_t)gac * K + ach;
  const __bf16* gB = Bt + (size_t)gbc * K + ach;

  v8f acc[2][4] = {};

  // issue one 128x32 + 128x32 tile (4 async b128 ops per thread)
  auto issue_tile = [&](int kt, int buf) {
    const int k0 = kt * BK;
    uint64_t a64 = (uint64_t)(uintptr_t)(gA + k0);
    uint64_t b64 = (uint64_t)(uintptr_t)(gB + k0);
    uint32_t la  = (uint32_t)(uintptr_t)&As[buf][arow][ach];
    uint32_t lb  = (uint32_t)(uintptr_t)&Bs[buf][arow][ach];
    // INST_OFFSET is applied to both the LDS and global sides.
    asm volatile("global_load_async_to_lds_b128 %0, %1, off"
                 :: "v"(la), "v"(a64) : "memory");
    asm volatile("global_load_async_to_lds_b128 %0, %1, off offset:16"
                 :: "v"(la), "v"(a64) : "memory");
    asm volatile("global_load_async_to_lds_b128 %0, %1, off"
                 :: "v"(lb), "v"(b64) : "memory");
    asm volatile("global_load_async_to_lds_b128 %0, %1, off offset:16"
                 :: "v"(lb), "v"(b64) : "memory");
  };

  const int ksteps = K / BK;
  int cur = 0;

  issue_tile(0, 0);                       // prologue: tile 0 in flight

  for (int kk = 0; kk < ksteps; ++kk) {
    __syncthreads();                      // prior reads of buf[cur^1] are done

    if (kk + 1 < ksteps) {
      issue_tile(kk + 1, cur ^ 1);        // overlap next transfer with compute
      asm volatile("s_wait_asynccnt 0x4" ::: "memory");   // tile kk complete
    } else {
      asm volatile("s_wait_asynccnt 0x0" ::: "memory");
    }
    if (kk + 2 < ksteps) {                // warm L2 for the tile after next
      __builtin_prefetch(gA + (kk + 2) * BK, 0, 0);
      __builtin_prefetch(gB + (kk + 2) * BK, 0, 0);
    }

    __syncthreads();                      // tile kk visible to all waves

    // ---- fragments per documented CDNA5 WMMA VGPR layouts ----
    union AV { v16bf v; u32x4 q[2]; };
    AV af[2], bfm[4];
#pragma unroll
    for (int tm = 0; tm < 2; ++tm) {
      const int m = wm + tm * 16 + lr;    // A: row m, chunks k=half*8, 16+half*8
      af[tm].q[0] = *(const u32x4*)&As[cur][m][half * 8];
      af[tm].q[1] = *(const u32x4*)&As[cur][m][16 + half * 8];
    }
#pragma unroll
    for (int tn = 0; tn < 4; ++tn) {
      const int n = wn + tn * 16 + lr;    // B (N-major): chunks k=half*8, 16+half*8
      bfm[tn].q[0] = *(const u32x4*)&Bs[cur][n][half * 8];
      bfm[tn].q[1] = *(const u32x4*)&Bs[cur][n][16 + half * 8];
    }

#pragma unroll
    for (int tm = 0; tm < 2; ++tm)
#pragma unroll
      for (int tn = 0; tn < 4; ++tn)
        acc[tm][tn] = __builtin_amdgcn_wmma_f32_16x16x32_bf16(
            /*neg_a=*/false, af[tm].v, /*neg_b=*/false, bfm[tn].v,
            /*c_mod=*/(short)0, acc[tm][tn],
            /*reuse_a=*/false, /*reuse_b=*/false);

    cur ^= 1;
  }

  // ---- epilogue: C/D layout -> M = r + half*8, N = lane%16 ----
#pragma unroll
  for (int tm = 0; tm < 2; ++tm)
#pragma unroll
    for (int tn = 0; tn < 4; ++tn) {
      const int col = n0 + wn + tn * 16 + lr;
      if (col >= Nt) continue;
#pragma unroll
      for (int r = 0; r < 8; ++r) {
        const int row = m0 + wm + tm * 16 + half * 8 + r;
        if (row < M) C[(size_t)row * Nt + col] = acc[tm][tn][r];
      }
    }
}

// ---------------------------------------------------------------------------
// Edge aggregation:  agg[dst,f] += coef[e] * h[src,f]   (bandwidth-bound stage)
// grid = (E, ceil(F/256)), block = 256
// ---------------------------------------------------------------------------
__global__ void k_agg(const float* __restrict__ h, const float* __restrict__ coef,
                      const int* __restrict__ ei, float* __restrict__ agg, int F) {
  const int e = blockIdx.x;
  const int f = blockIdx.y * blockDim.x + threadIdx.x;
  if (f >= F) return;
  const float c = coef[e];
  const int s = ei[e];
  const int d = ei[N_EDGES + e];
  const float v = __builtin_nontemporal_load(&h[(size_t)s * F + f]);
  atomicAdd(&agg[(size_t)d * F + f], c * v);
}

// z = agg + dinv2[row]*h + b[f]   (in place into the agg buffer)
__global__ void k_combine(float* __restrict__ z, const float* __restrict__ h,
                          const float* __restrict__ dinv2,
                          const float* __restrict__ b, int F, size_t total) {
  size_t i = (size_t)blockIdx.x * blockDim.x + threadIdx.x;
  if (i >= total) return;
  const int row = (int)(i / (size_t)F);
  const int f   = (int)(i - (size_t)row * F);
  z[i] = z[i] + dinv2[row] * h[i] + b[f];
}

// BN column statistics: partial sums over row chunks, atomically reduced
__global__ void k_bnstats(const float* __restrict__ z, float* __restrict__ s,
                          float* __restrict__ s2, int F, int rows_per) {
  const int f = blockIdx.x * blockDim.x + threadIdx.x;
  if (f >= F) return;
  int r0 = blockIdx.y * rows_per;
  int r1 = r0 + rows_per;
  if (r1 > N_NODES) r1 = N_NODES;
  float a = 0.f, b = 0.f;
  for (int r = r0; r < r1; ++r) {
    const float v = z[(size_t)r * F + f];
    a += v;
    b += v * v;
  }
  atomicAdd(&s[f], a);
  atomicAdd(&s2[f], b);
}

// turn (sum, sumsq) into (scale, shift):  y = z*scale + shift
__global__ void k_bnfin(float* __restrict__ s, float* __restrict__ s2,
                        const float* __restrict__ g, const float* __restrict__ be,
                        int F) {
  const int f = blockIdx.x * blockDim.x + threadIdx.x;
  if (f >= F) return;
  const float m   = s[f] * (1.0f / (float)N_NODES);
  const float var = s2[f] * (1.0f / (float)N_NODES) - m * m;
  const float inv = rsqrtf(var + EPSV);
  const float sc  = inv * g[f];
  s[f]  = sc;
  s2[f] = be[f] - m * sc;
}

// y = bf16(relu(z*scale + shift)) -> becomes next layer's A matrix
__global__ void k_bnrelu_cvt(const float* __restrict__ z, const float* __restrict__ sc,
                             const float* __restrict__ sh, __bf16* __restrict__ out,
                             int F, size_t total) {
  size_t i = (size_t)blockIdx.x * blockDim.x + threadIdx.x;
  if (i >= total) return;
  const int f = (int)(i % (size_t)F);
  float v = z[i] * sc[f] + sh[f];
  v = v > 0.f ? v : 0.f;
  out[i] = (__bf16)v;
}

// ---------------------------------------------------------------------------
// Host-side orchestration
// ---------------------------------------------------------------------------
static inline int idiv_up(int a, int b) { return (a + b - 1) / b; }

extern "C" void kernel_launch(void* const* d_in, const int* in_sizes, int n_in,
                              void* d_out, int out_size, void* d_ws, size_t ws_size,
                              hipStream_t stream) {
  (void)in_sizes; (void)n_in; (void)out_size; (void)ws_size;

  const float* x   = (const float*)d_in[0];
  const int*   ei  = (const int*)  d_in[1];
  const float* ea  = (const float*)d_in[2];
  // d_in[3] = y (unused)
  const float* W1  = (const float*)d_in[4];
  const float* b1  = (const float*)d_in[5];
  const float* g1  = (const float*)d_in[6];
  const float* be1 = (const float*)d_in[7];
  const float* W2  = (const float*)d_in[8];
  const float* b2  = (const float*)d_in[9];
  const float* g2  = (const float*)d_in[10];
  const float* be2 = (const float*)d_in[11];
  const float* W3  = (const float*)d_in[12];
  const float* b3  = (const float*)d_in[13];
  float* out = (float*)d_out;

  // ---- workspace layout (256B aligned slabs) ----
  char* ws = (char*)d_ws;
  size_t off = 0;
  auto alloc = [&](size_t bytes) -> void* {
    off = (off + 255) & ~(size_t)255;
    void* p = ws + off;
    off += bytes;
    return p;
  };
  float*  coef   = (float*)alloc((size_t)N_EDGES * 4);
  float*  deg    = (float*)alloc((size_t)N_NODES * 4);
  float*  dinv   = (float*)alloc((size_t)N_NODES * 4);
  float*  dinv2  = (float*)alloc((size_t)N_NODES * 4);
  float*  bns    = (float*)alloc((size_t)DIM_H2 * 4);
  float*  bns2   = (float*)alloc((size_t)DIM_H2 * 4);
  __bf16* W1t    = (__bf16*)alloc((size_t)DIM_IN * DIM_H1 * 2);
  __bf16* W2t    = (__bf16*)alloc((size_t)DIM_H1 * DIM_H2 * 2);
  __bf16* W3t    = (__bf16*)alloc((size_t)DIM_H2 * DIM_OUT * 2);
  __bf16* actb   = (__bf16*)alloc((size_t)N_NODES * DIM_H2 * 2);  // bf16 activations
  float*  hbuf   = (float*)alloc((size_t)N_NODES * DIM_OUT * 4);  // GEMM output
  float*  aggbuf = (float*)alloc((size_t)N_NODES * DIM_H2 * 4);   // agg/z buffer

  const int TB = 256;

  // ---- graph prep ----
  hipMemsetAsync(deg, 0, (size_t)N_NODES * 4, stream);
  k_deg <<<idiv_up(N_EDGES, TB), TB, 0, stream>>>(ei, ea, deg);
  k_dinv<<<idiv_up(N_NODES, TB), TB, 0, stream>>>(deg, dinv, dinv2);
  k_coef<<<idiv_up(N_EDGES, TB), TB, 0, stream>>>(ei, ea, dinv, coef);

  // ---- weights -> transposed bf16 ----
  k_cvt_wt<<<idiv_up(DIM_IN * DIM_H1, TB), TB, 0, stream>>>(W1, W1t, DIM_IN, DIM_H1);
  k_cvt_wt<<<idiv_up(DIM_H1 * DIM_H2, TB), TB, 0, stream>>>(W2, W2t, DIM_H1, DIM_H2);
  k_cvt_wt<<<idiv_up(DIM_H2 * DIM_OUT, TB), TB, 0, stream>>>(W3, W3t, DIM_H2, DIM_OUT);

  // ---- x -> bf16 ----
  {
    size_t tot = (size_t)N_NODES * DIM_IN;
    k_cvt_act<<<(unsigned)((tot + TB - 1) / TB), TB, 0, stream>>>(x, actb, tot);
  }

  const int rows_per = 250;  // 200 row chunks for BN stats
  const int nchunks  = idiv_up(N_NODES, rows_per);

  // =================== layer 1: 256 -> 512 ===================
  {
    const int F = DIM_H1, K = DIM_IN;
    dim3 gg(idiv_up(N_NODES, BM), idiv_up(F, BN));
    k_gemm_bf16<<<gg, TB, 0, stream>>>(actb, W1t, hbuf, N_NODES, F, K);

    size_t tot = (size_t)N_NODES * F;
    hipMemsetAsync(aggbuf, 0, tot * 4, stream);
    dim3 ag(N_EDGES, idiv_up(F, TB));
    k_agg<<<ag, TB, 0, stream>>>(hbuf, coef, ei, aggbuf, F);
    k_combine<<<(unsigned)((tot + TB - 1) / TB), TB, 0, stream>>>(aggbuf, hbuf, dinv2, b1, F, tot);

    hipMemsetAsync(bns,  0, (size_t)F * 4, stream);
    hipMemsetAsync(bns2, 0, (size_t)F * 4, stream);
    dim3 sg(idiv_up(F, TB), nchunks);
    k_bnstats<<<sg, TB, 0, stream>>>(aggbuf, bns, bns2, F, rows_per);
    k_bnfin<<<idiv_up(F, TB), TB, 0, stream>>>(bns, bns2, g1, be1, F);
    k_bnrelu_cvt<<<(unsigned)((tot + TB - 1) / TB), TB, 0, stream>>>(aggbuf, bns, bns2, actb, F, tot);
  }

  // =================== layer 2: 512 -> 1024 ===================
  {
    const int F = DIM_H2, K = DIM_H1;
    dim3 gg(idiv_up(N_NODES, BM), idiv_up(F, BN));
    k_gemm_bf16<<<gg, TB, 0, stream>>>(actb, W2t, hbuf, N_NODES, F, K);

    size_t tot = (size_t)N_NODES * F;
    hipMemsetAsync(aggbuf, 0, tot * 4, stream);
    dim3 ag(N_EDGES, idiv_up(F, TB));
    k_agg<<<ag, TB, 0, stream>>>(hbuf, coef, ei, aggbuf, F);
    k_combine<<<(unsigned)((tot + TB - 1) / TB), TB, 0, stream>>>(aggbuf, hbuf, dinv2, b2, F, tot);

    hipMemsetAsync(bns,  0, (size_t)F * 4, stream);
    hipMemsetAsync(bns2, 0, (size_t)F * 4, stream);
    dim3 sg(idiv_up(F, TB), nchunks);
    k_bnstats<<<sg, TB, 0, stream>>>(aggbuf, bns, bns2, F, rows_per);
    k_bnfin<<<idiv_up(F, TB), TB, 0, stream>>>(bns, bns2, g2, be2, F);
    k_bnrelu_cvt<<<(unsigned)((tot + TB - 1) / TB), TB, 0, stream>>>(aggbuf, bns, bns2, actb, F, tot);
  }

  // =================== layer 3: 1024 -> 3000 (no BN/ReLU) ===================
  {
    const int F = DIM_OUT, K = DIM_H2;
    dim3 gg(idiv_up(N_NODES, BM), idiv_up(F, BN));
    k_gemm_bf16<<<gg, TB, 0, stream>>>(actb, W3t, hbuf, N_NODES, F, K);

    size_t tot = (size_t)N_NODES * F;
    hipMemsetAsync(out, 0, tot * 4, stream);
    dim3 ag(N_EDGES, idiv_up(F, TB));
    k_agg<<<ag, TB, 0, stream>>>(hbuf, coef, ei, out, F);
    k_combine<<<(unsigned)((tot + TB - 1) / TB), TB, 0, stream>>>(out, hbuf, dinv2, b3, F, tot);
  }
}
